// SourceStateTransformer_40922448396567
// MI455X (gfx1250) — matrix operand order repaired and checked by reading, not compiled, not run on hardware
//
#include <hip/hip_runtime.h>
#include <math.h>

#define BB 8
#define CC 64
#define TT 1000
#define SS 200
#define DD 128
#define HH 4
#define HDIM 32
#define PP 25
#define STRD 10
#define NDEPTH 6
#define NCLS 4
#define NTOK 98            // (TT-PP)/STRD + 1
#define EPSLN 1e-5f

#define F_BIAS 1
#define F_GELU 2
#define F_RES  4

typedef __attribute__((ext_vector_type(16))) _Float16 v16h;
typedef __attribute__((ext_vector_type(8)))  float    v8f;

__device__ __forceinline__ float gelu_exact(float v) {
  return 0.5f * v * (1.0f + erff(v * 0.70710678118654752440f));
}

__device__ __forceinline__ unsigned pack_f16x2(float a, float b) {
  union { _Float16 h[2]; unsigned u; } p;
  p.h[0] = (_Float16)a;
  p.h[1] = (_Float16)b;
  return p.u;
}

// ---------------------------------------------------------------------------
// Generic batched GEMM: C[M,N] = A[M,K] @ B[K,N] (+bias)(gelu)(+Res)
// fp32 in/out, f16 WMMA compute with fp32 accumulate.
//
// Block = 128 threads = 4 waves; block tile 64x64; K-step 32.
// LDS tiles are stored pre-swizzled into WMMA fragment order (ISA 7.12.2):
//   - each lane's v16h operand is 32 contiguous bytes -> 2x ds_load_b128
//   - tile fill packs K-adjacent fp32 pairs into one f16x2 ds_store_b32
// Double-buffered LDS + register prefetch: one barrier per K-step; the next
// tile's global loads are issued before the 4 WMMAs of the current step.
// ---------------------------------------------------------------------------
__launch_bounds__(128)
__global__ void gemm_wmma_kernel(
    const float* __restrict__ A, long long sA, int lda,
    const float* __restrict__ Bw, long long sB, int ldb,
    float* __restrict__ C, long long sC, int ldc,
    const float* __restrict__ bias,
    const float* __restrict__ Res,
    int M, int N, int K, int flags)
{
  // [buf][4 subtiles * 512 halves]; fragment f, lane l at halves f*512 + l*16
  __shared__ _Float16 FragA[2][2048];
  __shared__ _Float16 FragB[2][2048];

  const int tid  = threadIdx.x;
  const int lane = tid & 31;
  const int wave = tid >> 5;
  const int wm = (wave >> 1) * 32;
  const int wn = (wave & 1) * 32;
  const int m0 = blockIdx.y * 64;
  const int n0 = blockIdx.x * 64;
  const long long bz = blockIdx.z;
  A  += bz * sA;
  Bw += bz * sB;
  C  += bz * sC;
  const float* Rp = Res ? (Res + bz * sC) : Res;

  // ---- per-thread cooperative fill plan (constant across K-steps) ----
  // A tile 64x32 viewed as 1024 K-pairs: pair p -> row m=p>>4, k=2*(p&15)
  // B tile 32x64 viewed as 1024 K-pairs: pair p -> col n=p&63, k=2*(p>>6)
  int amr[8], akc[8], bkr[8], bnc[8], offA[8], offB[8];
  #pragma unroll
  for (int i = 0; i < 8; ++i) {
    int p = i * 128 + tid;
    int m = p >> 4;
    int k = (p & 15) * 2;
    amr[i] = m; akc[i] = k;
    offA[i] = ((m >> 4) * 512) +
              (((m & 15) + (((k & 15) >= 8) ? 16 : 0)) * 16) +
              ((k >= 16) ? 8 : 0) + (k & 7);
    int n  = p & 63;
    int kq = (p >> 6) * 2;
    bnc[i] = n; bkr[i] = kq;
    offB[i] = ((n >> 4) * 512) +
              (((n & 15) + (((kq & 15) >= 8) ? 16 : 0)) * 16) +
              ((kq >= 16) ? 8 : 0) + (kq & 7);
  }

  float2 ra[8], rb[8];

  auto gload = [&](int k0) {
    #pragma unroll
    for (int i = 0; i < 8; ++i) {
      // A: aligned float2 of K-adjacent pair (K is always even)
      int gm = m0 + amr[i], gk = k0 + akc[i];
      const float* src = A + (long long)(gm < M ? gm : (M - 1)) * lda
                           + (gk < K ? gk : 0);
      float2 v = *(const float2*)src;
      if (gm >= M || gk >= K) { v.x = 0.0f; v.y = 0.0f; }
      ra[i] = v;
      // B: two coalesced b32 loads, ldb apart
      int gkb = k0 + bkr[i], gn = n0 + bnc[i];
      const float* sb = Bw + (long long)(gkb < K ? gkb : 0) * ldb
                           + (gn < N ? gn : 0);
      float w0 = sb[0];
      float w1 = sb[ldb];
      if (gkb >= K || gn >= N) { w0 = 0.0f; w1 = 0.0f; }
      rb[i].x = w0; rb[i].y = w1;
    }
  };

  auto sstore = [&](int buf) {
    #pragma unroll
    for (int i = 0; i < 8; ++i) {
      *(unsigned*)&FragA[buf][offA[i]] = pack_f16x2(ra[i].x, ra[i].y);
      *(unsigned*)&FragB[buf][offB[i]] = pack_f16x2(rb[i].x, rb[i].y);
    }
  };

  v8f acc00 = {}, acc01 = {}, acc10 = {}, acc11 = {};
  const int fa = (wave >> 1) * 2;    // A subtiles: rows wm..wm+15, wm+16..wm+31
  const int fb = (wave & 1) * 2;     // B subtiles: cols wn..wn+15, wn+16..wn+31

  gload(0);
  sstore(0);
  int cur = 0;
  for (int k0 = 0; k0 < K; k0 += 32) {
    __syncthreads();
    const bool more = (k0 + 32 < K);
    if (more) gload(k0 + 32);        // prefetch next tile into registers
    v16h a0 = *(const v16h*)&FragA[cur][(fa + 0) * 512 + lane * 16];
    v16h a1 = *(const v16h*)&FragA[cur][(fa + 1) * 512 + lane * 16];
    v16h b0 = *(const v16h*)&FragB[cur][(fb + 0) * 512 + lane * 16];
    v16h b1 = *(const v16h*)&FragB[cur][(fb + 1) * 512 + lane * 16];
    acc00 = __builtin_amdgcn_wmma_f32_16x16x32_f16(false, a0, false, b0, (short)0, acc00, false, false);
    acc01 = __builtin_amdgcn_wmma_f32_16x16x32_f16(false, a0, false, b1, (short)0, acc01, false, false);
    acc10 = __builtin_amdgcn_wmma_f32_16x16x32_f16(false, a1, false, b0, (short)0, acc10, false, false);
    acc11 = __builtin_amdgcn_wmma_f32_16x16x32_f16(false, a1, false, b1, (short)0, acc11, false, false);
    if (more) sstore(cur ^ 1);       // fill the other buffer (no barrier needed)
    cur ^= 1;
  }

  // epilogue (C/D layout: VGPR r -> M=r for lanes 0-15, M=8+r for lanes 16-31)
  const int nloc = lane & 15;
  const int mofs = (lane < 16) ? 0 : 8;
  auto store_tile = [&](const v8f& accv, int ti, int tj) {
    int gn = n0 + wn + tj * 16 + nloc;
    #pragma unroll
    for (int r = 0; r < 8; ++r) {
      int gm = m0 + wm + ti * 16 + mofs + r;
      if (gm < M && gn < N) {
        float v = accv[r];
        if (flags & F_BIAS) v += bias[gn];
        if (flags & F_GELU) v = gelu_exact(v);
        if (flags & F_RES)  v += Rp[(long long)gm * ldc + gn];
        C[(long long)gm * ldc + gn] = v;
      }
    }
  };
  store_tile(acc00, 0, 0);
  store_tile(acc01, 0, 1);
  store_tile(acc10, 1, 0);
  store_tile(acc11, 1, 1);
}

// ---------------------------------------------------------------------------
// LayerNorm over last dim (=128). One block (128 thr) per row.
// ---------------------------------------------------------------------------
__launch_bounds__(128)
__global__ void ln_kernel(const float* __restrict__ Xp, const float* __restrict__ g,
                          const float* __restrict__ bb, float* __restrict__ Yp)
{
  __shared__ float red[DD];
  const long long row = blockIdx.x;
  const int t = threadIdx.x;
  float v = Xp[row * DD + t];
  red[t] = v; __syncthreads();
  for (int s = DD / 2; s; s >>= 1) { if (t < s) red[t] += red[t + s]; __syncthreads(); }
  float mu = red[0] * (1.0f / DD); __syncthreads();
  float dv = v - mu;
  red[t] = dv * dv; __syncthreads();
  for (int s = DD / 2; s; s >>= 1) { if (t < s) red[t] += red[t + s]; __syncthreads(); }
  float var = red[0] * (1.0f / DD);
  Yp[row * DD + t] = dv * rsqrtf(var + EPSLN) * g[t] + bb[t];
}

// ---------------------------------------------------------------------------
// Tokenize: patch-projection (K=25) fused with LayerNorm. One block per token.
// ---------------------------------------------------------------------------
__launch_bounds__(128)
__global__ void tokenize_kernel(const float* __restrict__ x,
                                const float* __restrict__ pw, const float* __restrict__ pb,
                                const float* __restrict__ g, const float* __restrict__ bb,
                                float* __restrict__ out)
{
  __shared__ float xs[PP];
  __shared__ float red[DD];
  const long long row = blockIdx.x;                 // (b*NTOK + n)*CC + c
  const int t = threadIdx.x;
  const int c = (int)(row % CC);
  const long long bn = row / CC;
  const int nidx = (int)(bn % NTOK);
  const int b = (int)(bn / NTOK);
  if (t < PP) xs[t] = x[((long long)b * CC + c) * TT + nidx * STRD + t];
  __syncthreads();
  float v = pb[t];
  #pragma unroll
  for (int p = 0; p < PP; ++p) v += xs[p] * pw[p * DD + t];
  red[t] = v; __syncthreads();
  for (int s = DD / 2; s; s >>= 1) { if (t < s) red[t] += red[t + s]; __syncthreads(); }
  float mu = red[0] * (1.0f / DD); __syncthreads();
  float dv = v - mu;
  red[t] = dv * dv; __syncthreads();
  for (int s = DD / 2; s; s >>= 1) { if (t < s) red[t] += red[t + s]; __syncthreads(); }
  float var = red[0] * (1.0f / DD);
  out[row * DD + t] = dv * rsqrtf(var + EPSLN) * g[t] + bb[t];
}

// ---------------------------------------------------------------------------
// pos MLP: 3 -> 128 (gelu) -> 128, optional scale and additive input.
// ---------------------------------------------------------------------------
__launch_bounds__(128)
__global__ void posmlp_kernel(const float* __restrict__ pos,
                              const float* __restrict__ W1, const float* __restrict__ b1,
                              const float* __restrict__ W2, const float* __restrict__ b2,
                              const float* __restrict__ addin,
                              const float* __restrict__ scale_ptr,
                              float* __restrict__ out)
{
  __shared__ float h[DD];
  const int row = blockIdx.x, t = threadIdx.x;
  float p0 = pos[row * 3 + 0], p1 = pos[row * 3 + 1], p2 = pos[row * 3 + 2];
  float v = b1[t] + p0 * W1[t] + p1 * W1[DD + t] + p2 * W1[2 * DD + t];
  h[t] = gelu_exact(v);
  __syncthreads();
  float o = b2[t];
  #pragma unroll 8
  for (int e = 0; e < DD; ++e) o += h[e] * W2[e * DD + t];
  float sc = scale_ptr ? scale_ptr[0] : 1.0f;
  o *= sc;
  if (addin) o += addin[(long long)row * DD + t];
  out[(long long)row * DD + t] = o;
}

// ---------------------------------------------------------------------------
// Distance bias: out[b,q,k] = -||qpos - kpos||^2 / (2 sigma^2)
// ---------------------------------------------------------------------------
__global__ void bias_kernel(const float* __restrict__ qpos, int qPerB,
                            const float* __restrict__ kpos, int kPerB,
                            float* __restrict__ out, int qlen, int klen)
{
  const int b = blockIdx.y, q = blockIdx.x;
  const float* qp = qpos + ((qPerB ? (long long)b * qlen : 0) + q) * 3;
  const float qx = qp[0], qy = qp[1], qz = qp[2];
  const float inv = 1.0f / (2.0f * 0.35f * 0.35f);
  for (int k = threadIdx.x; k < klen; k += blockDim.x) {
    const float* kp = kpos + ((kPerB ? (long long)b * klen : 0) + k) * 3;
    float dx = qx - kp[0], dy = qy - kp[1], dz = qz - kp[2];
    out[((long long)b * qlen + q) * klen + k] = -(dx * dx + dy * dy + dz * dz) * inv;
  }
}

// ---------------------------------------------------------------------------
// sens = m * (tokens + pe) + (1-m) * mask_tok   (in place over tokens)
// ---------------------------------------------------------------------------
__global__ void sens_mask_kernel(float* sens, const float* __restrict__ sensPE,
                                 const float* __restrict__ maskf,
                                 const float* __restrict__ mask_tok)
{
  long long idx = (long long)blockIdx.x * blockDim.x + threadIdx.x;
  const long long total = (long long)BB * NTOK * CC * DD;
  if (idx >= total) return;
  int d = (int)(idx & (DD - 1));
  int c = (int)((idx >> 7) % CC);
  long long bn = idx / ((long long)CC * DD);
  int b = (int)(bn / NTOK);
  float m = maskf[b * CC + c];
  float v = sens[idx] + sensPE[((long long)b * CC + c) * DD + d];
  sens[idx] = m * v + (1.0f - m) * mask_tok[d];
}

__global__ void u8_to_f32_kernel(const unsigned char* __restrict__ in,
                                 float* __restrict__ out, int n)
{
  int i = blockIdx.x * blockDim.x + threadIdx.x;
  if (i < n) out[i] = in[i] ? 1.0f : 0.0f;
}

// ---------------------------------------------------------------------------
// Attention: one block per (h, n, b); K/V head slice staged in LDS;
// one query per thread, online softmax with distance bias + key mask.
// ---------------------------------------------------------------------------
__launch_bounds__(256)
__global__ void attn_kernel(const float* __restrict__ Q, long long qsb, long long qsn,
                            const float* __restrict__ K, const float* __restrict__ V,
                            const float* __restrict__ biasB,
                            const float* __restrict__ kmask,
                            float* __restrict__ Out,
                            int qlen, int klen)
{
  __shared__ float Ks[SS * HDIM];
  __shared__ float Vs[SS * HDIM];
  const int h = blockIdx.x, n = blockIdx.y, b = blockIdx.z;
  const int Nn = gridDim.y;
  const int tid = threadIdx.x;
  const long long kvbase = ((long long)(b * Nn + n) * klen) * DD + h * HDIM;
  for (int i = tid; i < klen * HDIM; i += blockDim.x) {
    int r = i / HDIM, cc = i % HDIM;
    Ks[i] = K[kvbase + (long long)r * DD + cc];
    Vs[i] = V[kvbase + (long long)r * DD + cc];
  }
  __syncthreads();
  const int q = tid;
  if (q >= qlen) return;
  const float* qp = Q + (long long)b * qsb + (long long)n * qsn + (long long)q * DD + h * HDIM;
  float qr[HDIM];
  #pragma unroll
  for (int e = 0; e < HDIM; ++e) qr[e] = qp[e];
  const float* brow = biasB + ((long long)b * qlen + q) * klen;
  const float* km = kmask ? (kmask + (long long)b * klen) : kmask;
  const float scale = 0.17677669529663687f;   // (D/H)^-0.5 = 32^-0.5
  float m = -3.0e38f, s = 0.0f;
  float acc[HDIM];
  #pragma unroll
  for (int e = 0; e < HDIM; ++e) acc[e] = 0.0f;
  for (int k = 0; k < klen; ++k) {
    float dot = 0.0f;
    #pragma unroll
    for (int e = 0; e < HDIM; ++e) dot += qr[e] * Ks[k * HDIM + e];
    float l = dot * scale + brow[k];
    if (km && km[k] == 0.0f) l = -10000.0f;
    float nm = fmaxf(m, l);
    float corr = expf(m - nm);
    float p = expf(l - nm);
    s = s * corr + p;
    #pragma unroll
    for (int e = 0; e < HDIM; ++e) acc[e] = acc[e] * corr + p * Vs[k * HDIM + e];
    m = nm;
  }
  float inv = 1.0f / s;
  float* op = Out + ((long long)(b * Nn + n) * qlen + q) * DD + h * HDIM;
  #pragma unroll
  for (int e = 0; e < HDIM; ++e) op[e] = acc[e] * inv;
}

// ---------------------------------------------------------------------------
// time_mix recurrence: one thread per (b,s,d), sequential over t (N=98).
// ---------------------------------------------------------------------------
__global__ void scan_kernel(const float* __restrict__ GC, float* __restrict__ YS)
{
  long long idx = (long long)blockIdx.x * blockDim.x + threadIdx.x;
  const long long total = (long long)BB * SS * DD;
  if (idx >= total) return;
  int d = (int)(idx % DD);
  int s = (int)((idx / DD) % SS);
  int b = (int)(idx / ((long long)DD * SS));
  float st = 0.0f;
  for (int t = 0; t < NTOK; ++t) {
    long long row = (long long)(b * NTOK + t) * SS + s;
    const float* g = GC + row * (2 * DD);
    float a = 1.0f / (1.0f + expf(-g[d]));
    float c = tanhf(g[DD + d]);
    st = (1.0f - a) * st + a * c;
    YS[row * DD + d] = st;
  }
}

// ---------------------------------------------------------------------------
// pooling: scores, softmax-per-batch, weighted sum
// ---------------------------------------------------------------------------
__launch_bounds__(128)
__global__ void rowdot_kernel(const float* __restrict__ Xp, const float* __restrict__ w,
                              const float* __restrict__ bptr, float* __restrict__ out)
{
  __shared__ float red[DD];
  const long long row = blockIdx.x;
  const int t = threadIdx.x;
  red[t] = Xp[row * DD + t] * w[t];
  __syncthreads();
  for (int s = DD / 2; s; s >>= 1) { if (t < s) red[t] += red[t + s]; __syncthreads(); }
  if (t == 0) out[row] = red[0] + bptr[0];
}

__launch_bounds__(256)
__global__ void softmax_rows_kernel(float* p, int NM)
{
  __shared__ float red[256];
  const int b = blockIdx.x, t = threadIdx.x;
  float* row = p + (long long)b * NM;
  float m = -3.0e38f;
  for (int i = t; i < NM; i += 256) m = fmaxf(m, row[i]);
  red[t] = m; __syncthreads();
  for (int s = 128; s; s >>= 1) { if (t < s) red[t] = fmaxf(red[t], red[t + s]); __syncthreads(); }
  m = red[0]; __syncthreads();
  float sum = 0.0f;
  for (int i = t; i < NM; i += 256) sum += expf(row[i] - m);
  red[t] = sum; __syncthreads();
  for (int s = 128; s; s >>= 1) { if (t < s) red[t] += red[t + s]; __syncthreads(); }
  float inv = 1.0f / red[0];
  __syncthreads();
  for (int i = t; i < NM; i += 256) row[i] = expf(row[i] - m) * inv;
}

__global__ void pool_kernel(const float* __restrict__ w, const float* __restrict__ Xp,
                            float* __restrict__ out, int NM)
{
  int idx = blockIdx.x * blockDim.x + threadIdx.x;
  if (idx >= BB * DD) return;
  int b = idx / DD, d = idx % DD;
  const float* wr = w + (long long)b * NM;
  const float* xr = Xp + (long long)b * NM * DD + d;
  float acc = 0.0f;
  for (int m = 0; m < NM; ++m) acc += wr[m] * xr[(long long)m * DD];
  out[idx] = acc;
}

// ---------------------------------------------------------------------------
// Classifier head: LN -> 128x128 gelu -> 128x4. One block per batch row.
// ---------------------------------------------------------------------------
__launch_bounds__(128)
__global__ void cls_kernel(const float* __restrict__ pooled,
                           const float* __restrict__ g, const float* __restrict__ bb,
                           const float* __restrict__ w1, const float* __restrict__ b1,
                           const float* __restrict__ w2, const float* __restrict__ b2,
                           float* __restrict__ logits)
{
  __shared__ float red[DD];
  __shared__ float buf[DD];
  const int b = blockIdx.x, t = threadIdx.x;
  float v = pooled[b * DD + t];
  red[t] = v; __syncthreads();
  for (int s = DD / 2; s; s >>= 1) { if (t < s) red[t] += red[t + s]; __syncthreads(); }
  float mu = red[0] * (1.0f / DD); __syncthreads();
  float dv = v - mu;
  red[t] = dv * dv; __syncthreads();
  for (int s = DD / 2; s; s >>= 1) { if (t < s) red[t] += red[t + s]; __syncthreads(); }
  float var = red[0] * (1.0f / DD); __syncthreads();
  buf[t] = dv * rsqrtf(var + EPSLN) * g[t] + bb[t];
  __syncthreads();
  float h = b1[t];
  for (int e = 0; e < DD; ++e) h += buf[e] * w1[e * DD + t];
  h = gelu_exact(h);
  __syncthreads();
  buf[t] = h;
  __syncthreads();
  if (t < NCLS) {
    float o = b2[t];
    for (int e = 0; e < DD; ++e) o += buf[e] * w2[e * NCLS + t];
    logits[b * NCLS + t] = o;
  }
}

__global__ void recon_mask_kernel(float* recon, const float* __restrict__ maskf)
{
  long long idx = (long long)blockIdx.x * blockDim.x + threadIdx.x;
  const long long total = (long long)BB * NTOK * CC * DD;
  if (idx >= total) return;
  int c = (int)((idx >> 7) % CC);
  int b = (int)(idx / ((long long)NTOK * CC * DD));
  recon[idx] *= maskf[b * CC + c];
}

// ===========================================================================
extern "C" void kernel_launch(void* const* d_in, const int* in_sizes, int n_in,
                              void* d_out, int out_size, void* d_ws, size_t ws_size,
                              hipStream_t stream)
{
  auto F = [&](int i) { return (const float*)d_in[i]; };
  const float* x          = F(0);
  const float* sensor_pos = F(1);
  const unsigned char* sensor_mask = (const unsigned char*)d_in[2];
  const float* source_pos = F(3);
  const float* adjacency  = F(4);

  // params, flattened in setup_inputs()/make_params() insertion order
  const float* tok_w = F(5), *tok_b = F(6), *tokn_g = F(7), *tokn_b = F(8);
  const float* spe1w = F(9), *spe1b = F(10), *spe2w = F(11), *spe2b = F(12);
  const float* qpe1w = F(13), *qpe1b = F(14), *qpe2w = F(15), *qpe2b = F(16);
  const float* mask_tok = F(17);
  const float* src_q = F(18);
  const float* s2s_qw = F(19), *s2s_qb = F(20), *s2s_kw = F(21), *s2s_kb = F(22);
  const float* s2s_vw = F(23), *s2s_vb = F(24), *s2s_ow = F(25), *s2s_ob = F(26);
  const int BLK0 = 27;                       // 6 blocks x 16 leaves
  const float* srcn_g = F(123), *srcn_b = F(124);
  const float* pool_w = F(125), *pool_b = F(126);
  const float* clsln_g = F(127), *clsln_b = F(128);
  const float* cls1w = F(129), *cls1b = F(130), *cls2w = F(131), *cls2b = F(132);
  const float* rpe1w = F(133), *rpe1b = F(134), *rpe2w = F(135), *rpe2b = F(136);
  const float* rec_scale = F(137);
  const float* rq_w = F(138), *rq_b = F(139), *rk_w = F(140), *rk_b = F(141);
  const float* rv_w = F(142), *rv_b = F(143), *ro_w = F(144), *ro_b = F(145);

  const long long MbnS = (long long)BB * NTOK * SS;   // 156800
  const long long MbnC = (long long)BB * NTOK * CC;   // 50176

  // ---- workspace layout (floats) ----
  float* ws = (float*)d_ws;
  size_t off = 0;
  auto alloc = [&](size_t n) { float* p = ws + off; off += n; return p; };
  float* maskf    = alloc((size_t)BB * CC);
  float* sensPE   = alloc((size_t)BB * CC * DD);
  float* recPE    = alloc((size_t)BB * CC * DD);
  float* qbase    = alloc((size_t)SS * DD);
  float* Qp       = alloc((size_t)SS * DD);
  float* Qr       = alloc((size_t)BB * CC * DD);
  float* bias_s2s = alloc((size_t)BB * SS * CC);
  float* bias_rec = alloc((size_t)BB * CC * SS);
  float* scores   = alloc((size_t)MbnS);
  float* pooled   = alloc((size_t)BB * DD);
  float* sensB    = alloc((size_t)MbnC * DD);        // tokens/sens; later recon attn-out
  float* Kp       = alloc((size_t)MbnC * DD);
  float* Vp       = alloc((size_t)MbnC * DD);
  float* AO       = alloc((size_t)MbnS * DD);        // s2s attn-out; later recon K; also "mixed"
  float* Xb       = alloc((size_t)MbnS * DD);        // main residual stream
  float* Yb       = alloc((size_t)MbnS * DD);        // LN / scan scratch; later srcn
  float* Wwide    = alloc((size_t)MbnS * 512);       // GC / FF hidden; later recon V
  float* KpR = AO; float* VpR = Wwide; float* AOrec = sensB;

  float* logits_out = (float*)d_out;
  float* recon_out  = (float*)d_out + BB * NCLS;

  auto gemm = [&](const float* A, long long sA, int lda,
                  const float* Bw, long long sB, int ldb,
                  float* Cc, long long sC, int ldc,
                  const float* bias, const float* Res,
                  int M, int Nn, int Kk, int batch, int flags) {
    dim3 g((Nn + 63) / 64, (M + 63) / 64, batch);
    gemm_wmma_kernel<<<g, dim3(128), 0, stream>>>(A, sA, lda, Bw, sB, ldb,
                                                  Cc, sC, ldc, bias, Res, M, Nn, Kk, flags);
  };

  // ---- stage 0: masks, positional encodings, biases ----
  u8_to_f32_kernel<<<(BB * CC + 255) / 256, 256, 0, stream>>>(sensor_mask, maskf, BB * CC);
  posmlp_kernel<<<BB * CC, DD, 0, stream>>>(sensor_pos, spe1w, spe1b, spe2w, spe2b,
                                            nullptr, nullptr, sensPE);
  posmlp_kernel<<<SS, DD, 0, stream>>>(source_pos, qpe1w, qpe1b, qpe2w, qpe2b,
                                       src_q, nullptr, qbase);
  posmlp_kernel<<<BB * CC, DD, 0, stream>>>(sensor_pos, rpe1w, rpe1b, rpe2w, rpe2b,
                                            nullptr, rec_scale, recPE);
  bias_kernel<<<dim3(SS, BB), 256, 0, stream>>>(source_pos, 0, sensor_pos, 1, bias_s2s, SS, CC);
  bias_kernel<<<dim3(CC, BB), 256, 0, stream>>>(sensor_pos, 1, source_pos, 0, bias_rec, CC, SS);

  // ---- stage 1: tokenize + sensor PE/mask ----
  tokenize_kernel<<<(unsigned)MbnC, DD, 0, stream>>>(x, tok_w, tok_b, tokn_g, tokn_b, sensB);
  sens_mask_kernel<<<(unsigned)((MbnC * DD + 255) / 256), 256, 0, stream>>>(sensB, sensPE, maskf, mask_tok);

  // ---- stage 2: source<-sensor cross attention ----
  gemm(qbase, 0, DD, s2s_qw, 0, DD, Qp, 0, DD, s2s_qb, nullptr, SS, DD, DD, 1, F_BIAS);
  gemm(sensB, 0, DD, s2s_kw, 0, DD, Kp, 0, DD, s2s_kb, nullptr, (int)MbnC, DD, DD, 1, F_BIAS);
  gemm(sensB, 0, DD, s2s_vw, 0, DD, Vp, 0, DD, s2s_vb, nullptr, (int)MbnC, DD, DD, 1, F_BIAS);
  attn_kernel<<<dim3(HH, NTOK, BB), 256, 0, stream>>>(Qp, 0, 0, Kp, Vp, bias_s2s, maskf,
                                                      AO, SS, CC);
  gemm(AO, 0, DD, s2s_ow, 0, DD, Xb, 0, DD, s2s_ob, nullptr, (int)MbnS, DD, DD, 1, F_BIAS);

  // ---- stage 3: transformer blocks ----
  for (int i = 0; i < NDEPTH; ++i) {
    const int pb = BLK0 + i * 16;
    const float* ng_g = F(pb + 0),  *ng_b = F(pb + 1);
    const float* nt_g = F(pb + 2),  *nt_b = F(pb + 3);
    const float* nf_g = F(pb + 4),  *nf_b = F(pb + 5);
    const float* gw   = F(pb + 6),  *gb   = F(pb + 7);
    const float* tiw  = F(pb + 8),  *tib  = F(pb + 9);
    const float* tow  = F(pb + 10), *tob  = F(pb + 11);
    const float* f1w  = F(pb + 12), *f1b  = F(pb + 13);
    const float* f2w  = F(pb + 14), *f2b  = F(pb + 15);

    // graph mix: x += (adj @ ln(x)) @ Wg + bg
    ln_kernel<<<(unsigned)MbnS, DD, 0, stream>>>(Xb, ng_g, ng_b, Yb);
    gemm(adjacency, 0, SS, Yb, (long long)SS * DD, DD, AO, (long long)SS * DD, DD,
         nullptr, nullptr, SS, DD, SS, BB * NTOK, 0);                     // mixed in AO
    gemm(AO, 0, DD, gw, 0, DD, Xb, 0, DD, gb, Xb, (int)MbnS, DD, DD, 1, F_BIAS | F_RES);

    // time mix: x += t_out(scan(t_in(ln(x))))
    ln_kernel<<<(unsigned)MbnS, DD, 0, stream>>>(Xb, nt_g, nt_b, Yb);
    gemm(Yb, 0, DD, tiw, 0, 2 * DD, Wwide, 0, 2 * DD, tib, nullptr,
         (int)MbnS, 2 * DD, DD, 1, F_BIAS);                               // GC in Wwide
    scan_kernel<<<(unsigned)(((long long)BB * SS * DD + 255) / 256), 256, 0, stream>>>(Wwide, Yb);
    gemm(Yb, 0, DD, tow, 0, DD, Xb, 0, DD, tob, Xb, (int)MbnS, DD, DD, 1, F_BIAS | F_RES);

    // feed-forward: x += ff2(gelu(ff1(ln(x))))
    ln_kernel<<<(unsigned)MbnS, DD, 0, stream>>>(Xb, nf_g, nf_b, Yb);
    gemm(Yb, 0, DD, f1w, 0, 4 * DD, Wwide, 0, 4 * DD, f1b, nullptr,
         (int)MbnS, 4 * DD, DD, 1, F_BIAS | F_GELU);                      // H in Wwide
    gemm(Wwide, 0, 4 * DD, f2w, 0, DD, Xb, 0, DD, f2b, Xb,
         (int)MbnS, DD, 4 * DD, 1, F_BIAS | F_RES);
  }

  // ---- stage 4: final LN, pooling, classifier ----
  ln_kernel<<<(unsigned)MbnS, DD, 0, stream>>>(Xb, srcn_g, srcn_b, Yb);   // srcn in Yb
  rowdot_kernel<<<(unsigned)MbnS, DD, 0, stream>>>(Yb, pool_w, pool_b, scores);
  softmax_rows_kernel<<<BB, 256, 0, stream>>>(scores, NTOK * SS);
  pool_kernel<<<(BB * DD + 255) / 256, 256, 0, stream>>>(scores, Yb, pooled, NTOK * SS);
  cls_kernel<<<BB, DD, 0, stream>>>(pooled, clsln_g, clsln_b, cls1w, cls1b, cls2w, cls2b,
                                    logits_out);

  // ---- stage 5: reconstruction cross attention ----
  gemm(recPE, 0, DD, rq_w, 0, DD, Qr, 0, DD, rq_b, nullptr, BB * CC, DD, DD, 1, F_BIAS);
  gemm(Yb, 0, DD, rk_w, 0, DD, KpR, 0, DD, rk_b, nullptr, (int)MbnS, DD, DD, 1, F_BIAS);
  gemm(Yb, 0, DD, rv_w, 0, DD, VpR, 0, DD, rv_b, nullptr, (int)MbnS, DD, DD, 1, F_BIAS);
  attn_kernel<<<dim3(HH, NTOK, BB), 256, 0, stream>>>(Qr, (long long)CC * DD, 0,
                                                      KpR, VpR, bias_rec, nullptr,
                                                      AOrec, CC, SS);
  gemm(AOrec, 0, DD, ro_w, 0, DD, recon_out, 0, DD, ro_b, nullptr,
       (int)MbnC, DD, DD, 1, F_BIAS);
  recon_mask_kernel<<<(unsigned)((MbnC * DD + 255) / 256), 256, 0, stream>>>(recon_out, maskf);
}
